// downsample_layer_82910048682467
// MI455X (gfx1250) — matrix operand order repaired and checked by reading, not compiled, not run on hardware
//
#include <hip/hip_runtime.h>
#include <hip/hip_bf16.h>

#define B_   32
#define CIN  64
#define COUT 128
#define H_   224
#define W_   224
#define HS   112
#define WS   112
#define HWS  (HS*WS)          // 12544
#define E_   4
#define SEH  16
#define BN_EPS 1e-5f
#define PIXB 128              // pixels per GEMM block (8 N-tiles)
#define NXBLK (HWS / PIXB)    // 98 pixel-blocks per batch
#define NBLK (B_ * NXBLK)     // 3136 GEMM blocks
#define LDSROW 72             // padded halfs per LDS pixel row (144B = 36-bank stride)

typedef _Float16 h8   __attribute__((ext_vector_type(8)));
typedef _Float16 v16h __attribute__((ext_vector_type(16)));
typedef float    v8f  __attribute__((ext_vector_type(8)));
typedef int      v4i  __attribute__((ext_vector_type(4)));

#if __has_builtin(__builtin_amdgcn_global_load_async_to_lds_b128)
#define HAVE_ASYNC_LDS 1
typedef __attribute__((address_space(1))) v4i gv4i;   // global int4
typedef __attribute__((address_space(3))) v4i lv4i;   // LDS int4 (32-bit ptr)
#endif

// -------- Kernel 1: GAP partials + pack strided slice to f16, K-contiguous --
// Block (hp, b) covers rows h=2hp, 2hp+1 for all 64 cin. Reads stay row-
// coalesced (this kernel owns the dominant 411 MB stream). The even row is
// transposed through LDS so xs is written as [b][p][i] (i contiguous) with one
// fully-coalesced contiguous 14,336 B store per block -> GEMM staging becomes
// a straight copy suitable for GLOBAL_LOAD_ASYNC_TO_LDS_B128.
__global__ void __launch_bounds__(256)
k_gap_pack_t(const float* __restrict__ x, float* __restrict__ gap_part,
             _Float16* __restrict__ xsh) {
  const int hp = blockIdx.x;          // 0..111  (h = 2*hp)
  const int b  = blockIdx.y;
  const int t  = threadIdx.x;
  const int c  = t >> 2;              // 0..63 cin
  const int q  = t & 3;               // quarter of the 224-wide row

  __shared__ _Float16 lt[WS * CIN];   // [ws][i] slab, 14336 B
  __shared__ float sred[256];

  const float* r0 = x + (((size_t)b * CIN + c) * H_ + 2 * hp) * W_ + q * 56;
  const float* r1 = r0 + W_;
  float s = 0.f;
#pragma unroll
  for (int j = 0; j < 28; ++j) {      // even row: gap + pack even-w elements
    float2 v = ((const float2*)r0)[j];
    s += v.x + v.y;
    lt[(q * 28 + j) * CIN + c] = (_Float16)v.x;
  }
#pragma unroll
  for (int j = 0; j < 14; ++j) {      // odd row: gap only
    float4 v = ((const float4*)r1)[j];
    s += (v.x + v.y) + (v.z + v.w);
  }
  sred[t] = s;
  __syncthreads();

  if (q == 0) {                       // deterministic per-(b,c,hp) partial
    const float tot = sred[4*c] + sred[4*c + 1] + sred[4*c + 2] + sred[4*c + 3];
    gap_part[((size_t)b * CIN + c) * HS + hp] = tot;
  }

  { // contiguous 14 KB store of the packed slab: xsh[b][hp*112 .. +111][0..63]
    uint4* gdst = (uint4*)(xsh + ((size_t)b * HWS + hp * WS) * CIN);
    const uint4* lsrc = (const uint4*)lt;
    for (int idx = t; idx < WS * CIN * 2 / 16; idx += 256) gdst[idx] = lsrc[idx];
  }
}

// -------- Kernel 1b: deterministic gap reduction over the 112 hp partials ---
__global__ void __launch_bounds__(128)
k_gap_reduce(const float* __restrict__ gap_part, float* __restrict__ gap) {
  const int c = blockIdx.x, b = blockIdx.y, t = threadIdx.x;
  __shared__ float red[128];
  const float* p = gap_part + ((size_t)b * CIN + c) * HS;
  red[t] = (t < HS) ? p[t] : 0.f;
  __syncthreads();
  for (int off = 64; off; off >>= 1) {
    if (t < off) red[t] += red[t + off];
    __syncthreads();
  }
  if (!t) gap[b * CIN + c] = red[0] * (1.f / (float)(H_ * W_));
}

// ------- Kernel 2: routing softmax + SE + fused per-batch f16 kernels -------
// K'[b][o][i] = se[b][o] * sum_e r[b][e] * expert_w[e][o][i]   (f16)
__global__ void __launch_bounds__(256)
k_route_se(const float* __restrict__ gap, const float* __restrict__ expert_w,
           const float* __restrict__ route_w, const float* __restrict__ route_b,
           const float* __restrict__ se_w1, const float* __restrict__ se_b1,
           const float* __restrict__ se_w2, const float* __restrict__ se_b2,
           _Float16* __restrict__ Kp) {
  const int b = blockIdx.x, t = threadIdx.x;
  __shared__ float g[CIN], lg[E_], r[E_], hh[SEH], se[COUT];
  if (t < CIN) g[t] = gap[b * CIN + t];
  __syncthreads();
  if (t < E_) {
    float a = route_b[t];
    for (int i = 0; i < CIN; ++i) a += g[i] * route_w[t * CIN + i];
    lg[t] = a;
  }
  if (t < SEH) {
    float a = se_b1[t];
    for (int i = 0; i < CIN; ++i) a += g[i] * se_w1[t * CIN + i];
    hh[t] = fmaxf(a, 0.f);
  }
  __syncthreads();
  if (t == 0) {
    float m = lg[0];
    for (int e = 1; e < E_; ++e) m = fmaxf(m, lg[e]);
    float s2 = 0.f;
    for (int e = 0; e < E_; ++e) { r[e] = __expf(lg[e] - m); s2 += r[e]; }
    for (int e = 0; e < E_; ++e) r[e] /= s2;
  }
  if (t < COUT) {
    float a = se_b2[t];
    for (int j = 0; j < SEH; ++j) a += hh[j] * se_w2[t * SEH + j];
    se[t] = 1.f / (1.f + __expf(-a));
  }
  __syncthreads();
  for (int idx = t; idx < COUT * CIN; idx += 256) {
    const int o = idx >> 6, i = idx & 63;
    float a = 0.f;
    for (int e = 0; e < E_; ++e) a += r[e] * expert_w[(e * COUT + o) * CIN + i];
    Kp[(size_t)b * (COUT * CIN) + idx] = (_Float16)(se[o] * a);
  }
}

// ------------- Kernel 3: WMMA GEMM y[b] = K'[b] @ xs[b] + BN partials -------
// 8 wave32s = 8 COUT tiles, 128 pixels. xs slab [b][p][i] is a contiguous
// 16 KB region -> staged with GLOBAL_LOAD_ASYNC_TO_LDS_B128 (ASYNCcnt), the
// per-lane LDS address inserting the 144 B padded row stride so the
// ds_load_b128 fragment reads are bank-conflict free.
__global__ void __launch_bounds__(256)
k_moe_conv(const _Float16* __restrict__ xsh, const _Float16* __restrict__ Kp,
           float* __restrict__ y, float* __restrict__ part) {
  const int b = blockIdx.y;
  const int p0 = blockIdx.x * PIXB;
  const int t = threadIdx.x;
  const int wave = t >> 5, lane = t & 31;
  const int lh = lane >> 4;   // K-half selector for fragments
  const int ll = lane & 15;   // M / N within tile

  __shared__ _Float16 lx[PIXB * LDSROW];

  const char* gsrc = (const char*)(xsh + ((size_t)b * HWS + p0) * CIN);
#if HAVE_ASYNC_LDS
  {
    const unsigned lbase = (unsigned)(uintptr_t)(&lx[0]);
#pragma unroll
    for (int k = 0; k < 4; ++k) {
      const int g = t + 256 * k;                     // 16B chunk index, 0..1023
      const unsigned loff = lbase + (unsigned)(g >> 3) * 144u + (unsigned)(g & 7) * 16u;
      __builtin_amdgcn_global_load_async_to_lds_b128(
          (gv4i*)(uintptr_t)(gsrc + (size_t)g * 16),
          (lv4i*)loff, 0, 0);
    }
#if __has_builtin(__builtin_amdgcn_s_wait_asynccnt)
    __builtin_amdgcn_s_wait_asynccnt(0);
#else
    asm volatile("s_wait_asynccnt 0x0" ::: "memory");
#endif
  }
#else
  { // fallback: b128 VGPR copy into the same padded layout
    for (int g = t; g < PIXB * 8; g += 256) {
      uint4 v = *(const uint4*)(gsrc + (size_t)g * 16);
      *(uint4*)((char*)lx + (g >> 3) * 144 + (g & 7) * 16) = v;
    }
  }
#endif
  __syncthreads();

  // A fragments (ISA 16-bit A 16x32 layout): per lane two 8-half chunks.
  const _Float16* arow = Kp + ((size_t)b * COUT + wave * 16 + ll) * CIN;
  h8 a0lo = *(const h8*)(arow +      8 * lh);
  h8 a0hi = *(const h8*)(arow + 16 + 8 * lh);
  h8 a1lo = *(const h8*)(arow + 32 + 8 * lh);
  h8 a1hi = *(const h8*)(arow + 48 + 8 * lh);
  v16h a0 = __builtin_shufflevector(a0lo, a0hi, 0,1,2,3,4,5,6,7,8,9,10,11,12,13,14,15);
  v16h a1 = __builtin_shufflevector(a1lo, a1hi, 0,1,2,3,4,5,6,7,8,9,10,11,12,13,14,15);

  float psum[8] = {0,0,0,0,0,0,0,0};
  float psq [8] = {0,0,0,0,0,0,0,0};

#pragma unroll
  for (int nt = 0; nt < PIXB / 16; ++nt) {
    const _Float16* brow = lx + (nt * 16 + ll) * LDSROW;
    h8 b0lo = *(const h8*)(brow + 8 * lh);
    h8 b0hi = *(const h8*)(brow + 16 + 8 * lh);
    h8 b1lo = *(const h8*)(brow + 32 + 8 * lh);
    h8 b1hi = *(const h8*)(brow + 48 + 8 * lh);
    v16h b0 = __builtin_shufflevector(b0lo, b0hi, 0,1,2,3,4,5,6,7,8,9,10,11,12,13,14,15);
    v16h b1 = __builtin_shufflevector(b1lo, b1hi, 0,1,2,3,4,5,6,7,8,9,10,11,12,13,14,15);

    v8f acc = {0.f,0.f,0.f,0.f,0.f,0.f,0.f,0.f};
    acc = __builtin_amdgcn_wmma_f32_16x16x32_f16(false, a0, false, b0,
                                                 (short)0, acc, false, false);
    acc = __builtin_amdgcn_wmma_f32_16x16x32_f16(false, a1, false, b1,
                                                 (short)0, acc, false, false);

    const int p = p0 + nt * 16 + ll;
#pragma unroll
    for (int rr = 0; rr < 8; ++rr) {
      const int o = wave * 16 + lh * 8 + rr;          // C layout: M = 8*lh + rr
      y[((size_t)b * COUT + o) * HWS + p] = acc[rr];
      psum[rr] += acc[rr];
      psq [rr] += acc[rr] * acc[rr];
    }
  }

  // Σy, Σy² across the 16 lanes sharing a channel -> deterministic partials
  float* bp = part + ((size_t)b * NXBLK + blockIdx.x) * (2 * COUT);
#pragma unroll
  for (int rr = 0; rr < 8; ++rr) {
    float s = psum[rr], qq = psq[rr];
    for (int m = 8; m; m >>= 1) {
      s  += __shfl_xor(s, m, 16);
      qq += __shfl_xor(qq, m, 16);
    }
    if (ll == 0) {
      const int o = wave * 16 + lh * 8 + rr;
      bp[o] = s;
      bp[COUT + o] = qq;
    }
  }
}

// -------- Kernel 3b: deterministic fixed-order reduction of BN partials -----
__global__ void __launch_bounds__(256)
k_stats_reduce(const float* __restrict__ part, float* __restrict__ sums) {
  const int idx = blockIdx.x;       // 0..255  (sum: 0..127, sumsq: 128..255)
  const int t = threadIdx.x;
  float s = 0.f;
  for (int j = t; j < NBLK; j += 256) s += part[(size_t)j * (2 * COUT) + idx];
  __shared__ float red[256];
  red[t] = s;
  __syncthreads();
  for (int off = 128; off; off >>= 1) {
    if (t < off) red[t] += red[t + off];
    __syncthreads();
  }
  if (!t) sums[idx] = red[0];
}

// -------------------- Kernel 4: fold BN stats into scale/shift --------------
__global__ void k_bn_finalize(const float* __restrict__ sums,
                              float* __restrict__ stats,
                              const float* __restrict__ gamma,
                              const float* __restrict__ beta) {
  const int c = threadIdx.x;
  const float n = (float)((size_t)B_ * HWS);
  const float mean = sums[c] / n;
  const float var = sums[COUT + c] / n - mean * mean;
  const float inv = gamma[c] * rsqrtf(var + BN_EPS);
  stats[c] = inv;                          // scale
  stats[COUT + c] = beta[c] - mean * inv;  // shift
}

// ------------------- Kernel 5: in-place float4 normalize --------------------
__global__ void __launch_bounds__(256)
k_bn_apply(float* __restrict__ y, const float* __restrict__ stats) {
  const size_t idx4 = (size_t)blockIdx.x * 256 + threadIdx.x;
  const size_t n4 = (size_t)B_ * COUT * HWS / 4;
  if (idx4 >= n4) return;
  float4 v = ((float4*)y)[idx4];
  const int c = (int)((idx4 * 4 / HWS) % COUT);    // HWS % 4 == 0 -> uniform c
  const float sc = stats[c], sh = stats[COUT + c];
  v.x = v.x * sc + sh; v.y = v.y * sc + sh;
  v.z = v.z * sc + sh; v.w = v.w * sc + sh;
  ((float4*)y)[idx4] = v;
}

extern "C" void kernel_launch(void* const* d_in, const int* in_sizes, int n_in,
                              void* d_out, int out_size, void* d_ws, size_t ws_size,
                              hipStream_t stream) {
  (void)in_sizes; (void)n_in; (void)out_size; (void)ws_size;
  const float* x        = (const float*)d_in[0];
  const float* expert_w = (const float*)d_in[1];
  const float* route_w  = (const float*)d_in[2];
  const float* route_b  = (const float*)d_in[3];
  const float* se_w1    = (const float*)d_in[4];
  const float* se_b1    = (const float*)d_in[5];
  const float* se_w2    = (const float*)d_in[6];
  const float* se_b2    = (const float*)d_in[7];
  const float* bn_gamma = (const float*)d_in[8];
  const float* bn_beta  = (const float*)d_in[9];
  float* y = (float*)d_out;

  // workspace carve-up (all 16B aligned)
  float*    gap_part = (float*)d_ws;                             // 229,376 f
  float*    gap      = gap_part + (size_t)B_ * CIN * HS;         //   2,048 f
  float*    part     = gap + B_ * CIN;                           // 802,816 f
  float*    sums     = part + (size_t)NBLK * 2 * COUT;           //     256 f
  float*    stats    = sums + 2 * COUT;                          //     256 f
  _Float16* Kp       = (_Float16*)(stats + 2 * COUT);            // 262,144 h
  _Float16* xsh      = Kp + (size_t)B_ * COUT * CIN;             // 25.7 M h

  k_gap_pack_t<<<dim3(HS, B_), 256, 0, stream>>>(x, gap_part, xsh);
  k_gap_reduce<<<dim3(CIN, B_), 128, 0, stream>>>(gap_part, gap);
  k_route_se<<<dim3(B_), 256, 0, stream>>>(gap, expert_w, route_w, route_b,
                                           se_w1, se_b1, se_w2, se_b2, Kp);
  k_moe_conv<<<dim3(NXBLK, B_), 256, 0, stream>>>(xsh, Kp, y, part);
  k_stats_reduce<<<dim3(2 * COUT), 256, 0, stream>>>(part, sums);
  k_bn_finalize<<<1, COUT, 0, stream>>>(sums, stats, bn_gamma, bn_beta);
  k_bn_apply<<<dim3((int)(((size_t)B_ * COUT * HWS / 4 + 255) / 256)), 256, 0, stream>>>(y, stats);
}
